// AlignOnlySubLayer_21818433864126
// MI455X (gfx1250) — matrix operand order repaired and checked by reading, not compiled, not run on hardware
//
#include <hip/hip_runtime.h>
#include <hip/hip_bf16.h>
#include <stdint.h>

typedef __attribute__((ext_vector_type(16))) __bf16 v16bf;
typedef __attribute__((ext_vector_type(8)))  float  v8f;

#define B_  8
#define L_  2048
#define D_  128

union BTile { v16bf v; uint4 q[2]; };

__device__ __forceinline__ unsigned short f2bf(float f){
  unsigned int u = __float_as_uint(f);
  u += 0x7FFFu + ((u >> 16) & 1u);   // round-to-nearest-even
  return (unsigned short)(u >> 16);
}

// ---------------- pass 0: f32 -> bf16 copies (+ transposed context) -------------
__global__ void cvt_kernel(const float* __restrict__ ctx, const float* __restrict__ mn,
                           unsigned short* __restrict__ ctxb,
                           unsigned short* __restrict__ ctxTb,
                           unsigned short* __restrict__ mainb){
  size_t i = (size_t)blockIdx.x * blockDim.x + threadIdx.x;   // B*L*D = 2^21 threads
  int d = (int)(i & (D_ - 1));
  int c = (int)((i >> 7) & (L_ - 1));
  int b = (int)(i >> 18);
  unsigned short cb = f2bf(ctx[i]);
  unsigned short mb = f2bf(mn[i]);
  ctxb[i]  = cb;
  mainb[i] = mb;
  ctxTb[((size_t)b * D_ + d) * L_ + c] = cb;   // ctxT[b][d][c]
}

// ---------------- pass 1: row stats (softmax over m, per c-row) -----------------
// Each wave owns 16 c-rows; block = 4 waves = 64 rows; grid = (2048/64, 8).
__global__ void __launch_bounds__(128)
pass1_stats(const unsigned short* __restrict__ ctxb,
            const unsigned short* __restrict__ mainb,
            float* __restrict__ rowmax, float* __restrict__ rowinv){
  const int lane = threadIdx.x & 31;
  const int wave = threadIdx.x >> 5;
  const int b   = blockIdx.y;
  const int c0  = blockIdx.x * 64 + wave * 16;
  const int h   = lane >> 4;      // half-wave select
  const int l15 = lane & 15;

  const unsigned short* ctx = ctxb  + (size_t)b * L_ * D_;
  const unsigned short* mn  = mainb + (size_t)b * L_ * D_;

  // A operand: 16 c-rows x 128 d, as 4 K-chunks of 32 (A-layout)
  BTile a[4];
  {
    const uint4* base = (const uint4*)(ctx + (size_t)(c0 + l15) * D_);
    #pragma unroll
    for (int k = 0; k < 4; ++k){ a[k].q[0] = base[4*k + h]; a[k].q[1] = base[4*k + h + 2]; }
  }

  float rm[8], rs[8];
  #pragma unroll
  for (int r = 0; r < 8; ++r){ rm[r] = -3.0e38f; rs[r] = 0.0f; }

  for (int mc = 0; mc < L_; mc += 128){           // 128 m-columns per chunk
    v8f acc[8];
    #pragma unroll
    for (int t = 0; t < 8; ++t) acc[t] = (v8f){0,0,0,0,0,0,0,0};
    #pragma unroll
    for (int k = 0; k < 4; ++k){
      #pragma unroll
      for (int t = 0; t < 8; ++t){
        const uint4* mb4 = (const uint4*)(mn + (size_t)(mc + 16*t + l15) * D_);
        BTile bt; bt.q[0] = mb4[4*k + h]; bt.q[1] = mb4[4*k + h + 2];
        acc[t] = __builtin_amdgcn_wmma_f32_16x16x32_bf16(false, a[k].v, false, bt.v,
                                                         (short)0, acc[t], false, false);
      }
    }
    // online softmax update; D layout: VGPR r -> row c0+r (lanes 0-15) / c0+8+r (lanes 16-31)
    #pragma unroll
    for (int r = 0; r < 8; ++r){
      float v = acc[0][r];
      #pragma unroll
      for (int t = 1; t < 8; ++t) v = fmaxf(v, acc[t][r]);
      #pragma unroll
      for (int mask = 1; mask <= 8; mask <<= 1) v = fmaxf(v, __shfl_xor(v, mask, 32));
      float nm = fmaxf(rm[r], v);
      float sc = __expf(rm[r] - nm);
      float s = 0.0f;
      #pragma unroll
      for (int t = 0; t < 8; ++t) s += __expf(acc[t][r] - nm);
      #pragma unroll
      for (int mask = 1; mask <= 8; mask <<= 1) s += __shfl_xor(s, mask, 32);
      rs[r] = rs[r] * sc + s;
      rm[r] = nm;
    }
  }
  if (l15 == 0){
    int rbase = b * L_ + c0 + h * 8;
    #pragma unroll
    for (int r = 0; r < 8; ++r){
      rowmax[rbase + r] = rm[r];
      rowinv[rbase + r] = 1.0f / rs[r];
    }
  }
}

// ---------------- pass 2: out = main - softmax(W)^T @ ctx -----------------------
// Each wave owns 16 m-rows x 128 d; block = 4 waves; grid = (2048/64, 8).
__global__ void __launch_bounds__(128)
pass2_out(const unsigned short* __restrict__ ctxb,
          const unsigned short* __restrict__ ctxTb,
          const unsigned short* __restrict__ mainb,
          const float* __restrict__ rowmax, const float* __restrict__ rowinv,
          const float* __restrict__ mainf, float* __restrict__ outp){
  __shared__ __align__(16) unsigned short smem[4 * 16 * 32];  // 1KB per wave: P^T tile
  const int lane = threadIdx.x & 31;
  const int wave = threadIdx.x >> 5;
  const int b   = blockIdx.y;
  const int m0  = blockIdx.x * 64 + wave * 16;
  const int h   = lane >> 4;
  const int l15 = lane & 15;

  const unsigned short* ctx  = ctxb  + (size_t)b * L_ * D_;
  const unsigned short* ctxT = ctxTb + (size_t)b * D_ * L_;
  const unsigned short* mn   = mainb + (size_t)b * L_ * D_;
  const float* rmax = rowmax + (size_t)b * L_;
  const float* rinv = rowinv + (size_t)b * L_;
  unsigned short* sp = smem + wave * (16 * 32);

  // A operand: 16 m-rows of main (register-resident for whole c sweep)
  BTile am[4];
  {
    const uint4* base = (const uint4*)(mn + (size_t)(m0 + l15) * D_);
    #pragma unroll
    for (int k = 0; k < 4; ++k){ am[k].q[0] = base[4*k + h]; am[k].q[1] = base[4*k + h + 2]; }
  }

  v8f outacc[8];
  #pragma unroll
  for (int t = 0; t < 8; ++t) outacc[t] = (v8f){0,0,0,0,0,0,0,0};

  for (int c0 = 0; c0 < L_; c0 += 32){
    // S^T[m, c] tiles for 32 c-columns, then P = exp(S^T - max[c]) * inv[c]
    #pragma unroll
    for (int ct = 0; ct < 2; ++ct){
      v8f s = (v8f){0,0,0,0,0,0,0,0};
      const uint4* cb4 = (const uint4*)(ctx + (size_t)(c0 + 16*ct + l15) * D_);
      #pragma unroll
      for (int k = 0; k < 4; ++k){
        BTile bt; bt.q[0] = cb4[4*k + h]; bt.q[1] = cb4[4*k + h + 2];
        s = __builtin_amdgcn_wmma_f32_16x16x32_bf16(false, am[k].v, false, bt.v,
                                                    (short)0, s, false, false);
      }
      int c = c0 + 16*ct + l15;            // this lane's c-column
      float mx  = rmax[c];
      float inv = rinv[c];
      #pragma unroll
      for (int r = 0; r < 8; ++r){
        float p = __expf(s[r] - mx) * inv;
        sp[(r + 8*h) * 32 + 16*ct + l15] = f2bf(p);   // row-major [m_local][c_local]
      }
    }
    // transpose D-layout -> A-layout through LDS (in-order DS within a wave)
    BTile a2;
    {
      const uint4* lb = (const uint4*)((const char*)sp + l15 * 64 + h * 16);
      a2.q[0] = lb[0];        // K 0..7  (lanes<16) / K 8..15  (lanes>=16)
      a2.q[1] = lb[2];        // K 16..23          / K 24..31
    }
    // out[m,d] += P^T (16m x 32c)  x  ctx (32c x 16d), B from transposed ctx copy
    #pragma unroll
    for (int t = 0; t < 8; ++t){
      const uint4* tb = (const uint4*)(ctxT + (size_t)(16*t + l15) * L_ + c0 + 8*h);
      BTile b2; b2.q[0] = tb[0]; b2.q[1] = tb[2];
      outacc[t] = __builtin_amdgcn_wmma_f32_16x16x32_bf16(false, a2.v, false, b2.v,
                                                          (short)0, outacc[t], false, false);
    }
  }

  // epilogue: out = main - weighted (f32)
  const float* mf = mainf + (size_t)b * L_ * D_;
  float*       op = outp  + (size_t)b * L_ * D_;
  #pragma unroll
  for (int t = 0; t < 8; ++t){
    int d = 16*t + l15;
    #pragma unroll
    for (int r = 0; r < 8; ++r){
      size_t idx = (size_t)(m0 + 8*h + r) * D_ + d;
      op[idx] = mf[idx] - outacc[t][r];
    }
  }
}

extern "C" void kernel_launch(void* const* d_in, const int* in_sizes, int n_in,
                              void* d_out, int out_size, void* d_ws, size_t ws_size,
                              hipStream_t stream){
  const float* ctx = (const float*)d_in[0];   // (8, 2048, 128) f32
  const float* mn  = (const float*)d_in[1];   // (8, 2048, 128) f32
  float* out = (float*)d_out;                 // (8, 2048, 128) f32
  char* ws = (char*)d_ws;

  // workspace layout (needs 12 MiB + 128 KiB)
  unsigned short* ctxb  = (unsigned short*)(ws);                        // 4 MiB
  unsigned short* mainb = (unsigned short*)(ws + (size_t)4*1024*1024);  // 4 MiB
  unsigned short* ctxTb = (unsigned short*)(ws + (size_t)8*1024*1024);  // 4 MiB
  float* rowmax = (float*)(ws + (size_t)12*1024*1024);                  // 64 KiB
  float* rowinv = (float*)(ws + (size_t)12*1024*1024 + 64*1024);        // 64 KiB

  size_t total = (size_t)B_ * L_ * D_;        // 2^21
  cvt_kernel <<<dim3((unsigned)(total / 256)), 256, 0, stream>>>(ctx, mn, ctxb, ctxTb, mainb);
  pass1_stats<<<dim3(L_ / 64, B_), 128, 0, stream>>>(ctxb, mainb, rowmax, rowinv);
  pass2_out  <<<dim3(L_ / 64, B_), 128, 0, stream>>>(ctxb, ctxTb, mainb, rowmax, rowinv, mn, out);
}
